// LAMEmbedder_70205535420767
// MI455X (gfx1250) — matrix operand order repaired and checked by reading, not compiled, not run on hardware
//
#include <hip/hip_runtime.h>
#include <hip/hip_bf16.h>
#include <math.h>

typedef __attribute__((ext_vector_type(16))) __bf16 v16bf;
typedef __attribute__((ext_vector_type(8)))  __bf16 v8bf;   // 16-byte chunk
typedef __attribute__((ext_vector_type(8)))  float  v8f;

#define BB  64
#define SS  256
#define DD  384
#define HH  12
#define NLAYER 6
#define DHH 32
#define DFF 1536
#define MM  (BB*SS)   // 16384 rows

// ------------------------- activations -------------------------
template<int ACT>
__device__ __forceinline__ float act_apply(float x) {
  if (ACT == 1) return x / (1.0f + __expf(-x));                              // SiLU
  if (ACT == 2) return 0.5f * x * (1.0f + erff(x * 0.7071067811865475f));    // exact GELU
  return x;
}

// Load a 16-element bf16 fragment from two 16-byte-aligned LDS chunks.
__device__ __forceinline__ v16bf frag16(const __bf16* p0, const __bf16* p1) {
  union { v16bf v; v8bf h[2]; } u;
  u.h[0] = *(const v8bf*)p0;
  u.h[1] = *(const v8bf*)p1;
  return u.v;
}

// ------------------------- WMMA GEMM ---------------------------
// C[M,N] = act(A[M,K] @ W[K,N] + bias), fp32 in/out, bf16 WMMA w/ f32 acc.
// K,N compile-time -> all strides fold into immediate offsets.
// Block tile 128x128, BK=32. 8 waves: 4 along M x 2 along N, wave tile 32x64
// (2x4 WMMA tiles -> 8 wmma per K-step vs 12 ds_load_b128).
// A tile in LDS row-major [m][k]; B tile in LDS TRANSPOSED [n][k] so both
// fragment types are contiguous 16B chunks -> ds_load_b128.
template<int ACT, int K, int N>
__global__ __launch_bounds__(256)
void gemm_bf16_wmma(const float* __restrict__ A, const float* __restrict__ W,
                    const float* __restrict__ bias, float* __restrict__ C)
{
  __shared__ __bf16 As[128][40];  // 128 rows x 32 k (pad to 40: 80B row, 16B aligned)
  __shared__ __bf16 Bs[128][40];  // 128 cols x 32 k (transposed tile)

  const int tid  = threadIdx.x;
  const int lane = tid & 31;
  const int wid  = tid >> 5;
  const int wm   = wid & 3;          // wave row group (0..3) -> rows wm*32
  const int wn   = wid >> 2;         // wave col group (0..1) -> cols wn*64
  const int bm0  = blockIdx.x * 128;
  const int bn0  = blockIdx.y * 128;

  v8f acc[2][4];
#pragma unroll
  for (int i = 0; i < 2; ++i)
#pragma unroll
    for (int j = 0; j < 4; ++j)
#pragma unroll
      for (int e = 0; e < 8; ++e) acc[i][j][e] = 0.0f;

  const int arow = tid >> 1, ac0 = (tid & 1) * 16;   // A fill: 2 thr/row, 16 floats
  const int bn   = tid & 127;                        // B fill: thread owns column n
  const int kb   = (tid >> 7) * 8;                   //   sub-blocks kb and kb+16
  const int hh   = lane >> 4;                        // lane half (K-half select)
  const int mn   = lane & 15;                        // M (for A) / N (for B,C) in tile

  const float* pa_base = A + (size_t)(bm0 + arow) * K + ac0;
  const float* pw_base = W + bn0 + bn;

  for (int k0 = 0; k0 < K; k0 += 32) {
    // ---- A: global row-major -> LDS row-major, fp32 -> bf16 ----
    const float4* pa = (const float4*)(pa_base + k0);
#pragma unroll
    for (int u = 0; u < 4; ++u) {
      float4 f = pa[u];
      As[arow][ac0 + 4*u + 0] = (__bf16)f.x;
      As[arow][ac0 + 4*u + 1] = (__bf16)f.y;
      As[arow][ac0 + 4*u + 2] = (__bf16)f.z;
      As[arow][ac0 + 4*u + 3] = (__bf16)f.w;
    }
    // ---- B: global row-major -> LDS transposed [n][k]; j*N is an immediate ----
#pragma unroll
    for (int sb = 0; sb < 2; ++sb) {
      const int kk0 = kb + sb * 16;
      const float* pw = pw_base + (size_t)(k0 + kk0) * N;
      union { v8bf v; __bf16 e[8]; } ub;
#pragma unroll
      for (int j = 0; j < 8; ++j)
        ub.e[j] = (__bf16)pw[j * N];               // coalesced across lanes per k-row
      *(v8bf*)&Bs[bn][kk0] = ub.v;                 // one ds_store_b128
    }
    if (k0 + 32 < K) {  // near-scope prefetch of next K tile
      __builtin_prefetch(pa_base + k0 + 32, 0, 3);
      __builtin_prefetch(pw_base + (size_t)(k0 + 32 + kb) * N, 0, 3);
    }
    __syncthreads();

    // ---- fragments: A lane m=mn, halves at k = 8h and 16+8h (contig 8 each);
    //      B lane n=mn, same: halves at k = 8h and 16+8h ----
    v16bf afrag[2], bfrag[4];
#pragma unroll
    for (int t = 0; t < 2; ++t) {
      const __bf16* ar = &As[wm*32 + t*16 + mn][0];
      afrag[t] = frag16(ar + 8*hh, ar + 16 + 8*hh);
    }
#pragma unroll
    for (int t = 0; t < 4; ++t) {
      const __bf16* br = &Bs[wn*64 + t*16 + mn][0];
      bfrag[t] = frag16(br + 8*hh, br + 16 + 8*hh);
    }
#pragma unroll
    for (int i = 0; i < 2; ++i)
#pragma unroll
      for (int j = 0; j < 4; ++j)
        acc[i][j] = __builtin_amdgcn_wmma_f32_16x16x32_bf16(
            false, afrag[i], false, bfrag[j], (short)0, acc[i][j], false, false);
    __syncthreads();
  }

  // ---- epilogue: C layout lane->(N=mn), vgpr e -> M = e + 8*hh ----
#pragma unroll
  for (int i = 0; i < 2; ++i) {
#pragma unroll
    for (int j = 0; j < 4; ++j) {
      int col = bn0 + wn*64 + j*16 + mn;
      float bv = bias[col];
#pragma unroll
      for (int e = 0; e < 8; ++e) {
        int row = bm0 + wm*32 + i*16 + 8*hh + e;
        C[(size_t)row * N + col] = act_apply<ACT>(acc[i][j][e] + bv);
      }
    }
  }
}

// ------------------------- small gate GEMM (N=12) -------------------------
__global__ __launch_bounds__(256)
void gemv_sigmoid(const float* __restrict__ X, const float* __restrict__ W,
                  const float* __restrict__ b, float* __restrict__ out)
{
  int idx = blockIdx.x * 256 + threadIdx.x;
  int r = idx / HH, hh = idx % HH;
  float s = b[hh];
  const float* xr = X + (size_t)r * DD;
  for (int k = 0; k < DD; ++k) s += xr[k] * W[(size_t)k * HH + hh];
  out[idx] = 1.0f / (1.0f + __expf(-s));
}

// ------------------------- per-head k normalization -------------------------
__global__ __launch_bounds__(256)
void knorm_kernel(float* __restrict__ Kb)
{
  int gid  = blockIdx.x * 8 + (threadIdx.x >> 5);   // (row*H + h), contiguous 32 elems
  int lane = threadIdx.x & 31;
  float v = Kb[(size_t)gid * 32 + lane];
  float s = v * v;
#pragma unroll
  for (int off = 16; off; off >>= 1) s += __shfl_xor(s, off, 32);
  Kb[(size_t)gid * 32 + lane] = v / (sqrtf(s) + 1e-6f);
}

// ------------------------- dual-timescale delta-rule scan -------------------------
// One wave per (b,h). Lane = output dim v; Sf/Ss columns live in VGPRs.
// Writes flux-gated output o = g*of + (1-g)*os in place over V.
__global__ __launch_bounds__(32)
void scan_kernel(const float* __restrict__ Q, const float* __restrict__ Kt,
                 float* __restrict__ Vt, const float* __restrict__ Beta,
                 const float* __restrict__ G, const int* __restrict__ Mask,
                 const float* __restrict__ dlogF, const float* __restrict__ dlogS)
{
  const int b = blockIdx.x / HH, hh = blockIdx.x % HH;
  const int lane = threadIdx.x;
  const float gf = 1.0f / (1.0f + __expf(-dlogF[hh]));
  const float gs = 1.0f / (1.0f + __expf(-dlogS[hh]));
  float Sf[32], Ssl[32];
#pragma unroll
  for (int k = 0; k < 32; ++k) { Sf[k] = 0.0f; Ssl[k] = 0.0f; }
  __shared__ float kb[32], qb[32];

  for (int s = 0; s < SS; ++s) {
    const int    tok  = b * SS + s;
    const size_t base = (size_t)tok * DD + hh * DHH;
    float kt = Kt[base + lane], qt = Q[base + lane], vt = Vt[base + lane];
    float bm = Beta[(size_t)tok * HH + hh] * (float)Mask[tok];
    float gg = G[(size_t)tok * HH + hh];
    kb[lane] = kt; qb[lane] = qt;
    __syncthreads();
    float pf = 0.0f, ps = 0.0f;
#pragma unroll
    for (int k = 0; k < 32; ++k) { pf += Sf[k] * kb[k]; ps += Ssl[k] * kb[k]; }
    float df  = bm * (vt - pf);
    float dsv = bm * (vt - ps);
    float of = 0.0f, os = 0.0f;
#pragma unroll
    for (int k = 0; k < 32; ++k) {
      Sf[k]  = gf * Sf[k]  + kb[k] * df;   of += Sf[k]  * qb[k];
      Ssl[k] = gs * Ssl[k] + kb[k] * dsv;  os += Ssl[k] * qb[k];
    }
    Vt[base + lane] = gg * of + (1.0f - gg) * os;
    __syncthreads();
  }
}

// ------------------------- embedding + LN -------------------------
__global__ __launch_bounds__(128)
void embed_ln_kernel(const int* __restrict__ ids, const float* __restrict__ we,
                     const float* __restrict__ pe, const float* __restrict__ te,
                     const float* __restrict__ g, const float* __restrict__ bb,
                     float* __restrict__ X)
{
  const int row = blockIdx.x;      // b*S + s
  const int s   = row % SS;
  const int t   = threadIdx.x;
  const int id  = ids[row];
  float vals[3]; float s1 = 0.0f, s2 = 0.0f;
#pragma unroll
  for (int u = 0; u < 3; ++u) {
    int d = t + u * 128;
    float x = we[(size_t)id * DD + d] + pe[(size_t)s * DD + d] + te[d];
    vals[u] = x; s1 += x; s2 += x * x;
  }
  __shared__ float r1[128], r2[128];
  r1[t] = s1; r2[t] = s2; __syncthreads();
  for (int off = 64; off; off >>= 1) {
    if (t < off) { r1[t] += r1[t + off]; r2[t] += r2[t + off]; }
    __syncthreads();
  }
  float mean = r1[0] / DD;
  float inv  = rsqrtf(r2[0] / DD - mean * mean + 1e-12f);
#pragma unroll
  for (int u = 0; u < 3; ++u) {
    int d = t + u * 128;
    X[(size_t)row * DD + d] = (vals[u] - mean) * inv * g[d] + bb[d];
  }
}

// ------------------------- residual add + LN (in place on X) -------------------------
__global__ __launch_bounds__(128)
void add_ln_kernel(float* __restrict__ X, const float* __restrict__ T,
                   const float* __restrict__ g, const float* __restrict__ bb)
{
  const int row = blockIdx.x;
  const int t   = threadIdx.x;
  float vals[3]; float s1 = 0.0f, s2 = 0.0f;
#pragma unroll
  for (int u = 0; u < 3; ++u) {
    int d = t + u * 128;
    float x = X[(size_t)row * DD + d] + T[(size_t)row * DD + d];
    vals[u] = x; s1 += x; s2 += x * x;
  }
  __shared__ float r1[128], r2[128];
  r1[t] = s1; r2[t] = s2; __syncthreads();
  for (int off = 64; off; off >>= 1) {
    if (t < off) { r1[t] += r1[t + off]; r2[t] += r2[t + off]; }
    __syncthreads();
  }
  float mean = r1[0] / DD;
  float inv  = rsqrtf(r2[0] / DD - mean * mean + 1e-12f);
#pragma unroll
  for (int u = 0; u < 3; ++u) {
    int d = t + u * 128;
    X[(size_t)row * DD + d] = (vals[u] - mean) * inv * g[d] + bb[d];
  }
}

// ------------------------- masked mean pooling -------------------------
__global__ __launch_bounds__(384)
void pool_kernel(const float* __restrict__ X, const int* __restrict__ Mask,
                 float* __restrict__ out)
{
  const int b = blockIdx.x, d = threadIdx.x;
  float s = 0.0f, ms = 0.0f;
  for (int t = 0; t < SS; ++t) {
    float m = (float)Mask[b * SS + t];
    s  += X[((size_t)(b * SS + t)) * DD + d] * m;
    ms += m;
  }
  out[(size_t)b * DD + d] = s / fmaxf(ms, 1e-9f);
}

// ------------------------- host orchestration -------------------------
extern "C" void kernel_launch(void* const* d_in, const int* in_sizes, int n_in,
                              void* d_out, int out_size, void* d_ws, size_t ws_size,
                              hipStream_t stream)
{
  const int*   ids   = (const int*)d_in[0];
  const int*   amask = (const int*)d_in[1];
  const float* we    = (const float*)d_in[2];
  const float* pe    = (const float*)d_in[3];
  const float* te    = (const float*)d_in[4];
  const float* elng  = (const float*)d_in[5];
  const float* elnb  = (const float*)d_in[6];
  const float* Wq = (const float*)d_in[7];   const float* bq = (const float*)d_in[8];
  const float* Wk = (const float*)d_in[9];   const float* bk = (const float*)d_in[10];
  const float* Wv = (const float*)d_in[11];  const float* bv = (const float*)d_in[12];
  const float* Wo = (const float*)d_in[13];  const float* bo = (const float*)d_in[14];
  const float* Wbe = (const float*)d_in[15]; const float* bbe = (const float*)d_in[16];
  const float* Wg = (const float*)d_in[17];  const float* bg = (const float*)d_in[18];
  const float* dlf = (const float*)d_in[19]; const float* dls = (const float*)d_in[20];
  const float* l1g = (const float*)d_in[21]; const float* l1b = (const float*)d_in[22];
  const float* fw1 = (const float*)d_in[23]; const float* fb1 = (const float*)d_in[24];
  const float* fw2 = (const float*)d_in[25]; const float* fb2 = (const float*)d_in[26];
  const float* l2g = (const float*)d_in[27]; const float* l2b = (const float*)d_in[28];

  float* ws = (float*)d_ws;
  float* X  = ws;
  float* Qb = X  + (size_t)MM * DD;
  float* Kb = Qb + (size_t)MM * DD;
  float* Vb = Kb + (size_t)MM * DD;
  float* Tb = Vb + (size_t)MM * DD;
  float* Hb = Tb + (size_t)MM * DD;         // FFN hidden, MM*DFF
  float* Bt = Hb + (size_t)MM * DFF;
  float* Gb = Bt + (size_t)MM * HH;

  embed_ln_kernel<<<MM, 128, 0, stream>>>(ids, we, pe, te, elng, elnb, X);

  dim3 gD(MM / 128, DD / 128);    // 128 x 3
  dim3 gF(MM / 128, DFF / 128);   // 128 x 12

  for (int l = 0; l < NLAYER; ++l) {
    const float* Wq_l = Wq + (size_t)l * DD * DD;  const float* bq_l = bq + (size_t)l * DD;
    const float* Wk_l = Wk + (size_t)l * DD * DD;  const float* bk_l = bk + (size_t)l * DD;
    const float* Wv_l = Wv + (size_t)l * DD * DD;  const float* bv_l = bv + (size_t)l * DD;
    const float* Wo_l = Wo + (size_t)l * DD * DD;  const float* bo_l = bo + (size_t)l * DD;
    const float* Wbe_l = Wbe + (size_t)l * DD * HH; const float* bbe_l = bbe + (size_t)l * HH;
    const float* Wg_l  = Wg  + (size_t)l * DD * HH; const float* bg_l  = bg  + (size_t)l * HH;
    const float* fw1_l = fw1 + (size_t)l * DD * DFF; const float* fb1_l = fb1 + (size_t)l * DFF;
    const float* fw2_l = fw2 + (size_t)l * DFF * DD; const float* fb2_l = fb2 + (size_t)l * DD;

    gemm_bf16_wmma<1, DD, DD><<<gD, 256, 0, stream>>>(X, Wq_l, bq_l, Qb);   // q = silu
    gemm_bf16_wmma<1, DD, DD><<<gD, 256, 0, stream>>>(X, Wk_l, bk_l, Kb);   // k = silu
    gemm_bf16_wmma<0, DD, DD><<<gD, 256, 0, stream>>>(X, Wv_l, bv_l, Vb);   // v linear
    knorm_kernel<<<(MM * HH) / 8, 256, 0, stream>>>(Kb);
    gemv_sigmoid<<<(MM * HH) / 256, 256, 0, stream>>>(X, Wbe_l, bbe_l, Bt);
    gemv_sigmoid<<<(MM * HH) / 256, 256, 0, stream>>>(X, Wg_l, bg_l, Gb);
    scan_kernel<<<BB * HH, 32, 0, stream>>>(Qb, Kb, Vb, Bt, Gb, amask,
                                            dlf + (size_t)l * HH, dls + (size_t)l * HH);
    gemm_bf16_wmma<0, DD, DD><<<gD, 256, 0, stream>>>(Vb, Wo_l, bo_l, Tb);  // out proj
    add_ln_kernel<<<MM, 128, 0, stream>>>(X, Tb, l1g + (size_t)l * DD, l1b + (size_t)l * DD);
    gemm_bf16_wmma<2, DD, DFF><<<gF, 256, 0, stream>>>(X, fw1_l, fb1_l, Hb); // gelu
    gemm_bf16_wmma<0, DFF, DD><<<gD, 256, 0, stream>>>(Hb, fw2_l, fb2_l, Tb);
    add_ln_kernel<<<MM, 128, 0, stream>>>(X, Tb, l2g + (size_t)l * DD, l2b + (size_t)l * DD);
  }

  pool_kernel<<<BB, 384, 0, stream>>>(X, amask, (float*)d_out);
}
